// myLSTM_24378234372576
// MI455X (gfx1250) — matrix operand order repaired and checked by reading, not compiled, not run on hardware
//
#include <hip/hip_runtime.h>
#include <hip/hip_bf16.h>
#include <math.h>

// Problem constants (match reference: B=256, H=1024, T_STEPS=128).
#define B_   256
#define H_   1024
#define T_   128
#define KDIM 2048   // [h | x] concatenated along K
#define NDIM 4096   // 4 gates (f,i,c,o) * H along N

// Fused step tiling: block = 4 waves (128 thr), M tile 64, per-gate N tile 16.
// B tile rows rr = gate*16 + j  ->  Wcat row gate*1024 + n0 + j.
#define BM   64
#define BN   64     // 4 gates x 16 columns
#define KT   32
#define LDA  40     // padded LDS row stride in bf16 elements (32 data + 8 pad)

typedef __attribute__((ext_vector_type(16))) __bf16 v16bf;
typedef __attribute__((ext_vector_type(8)))  __bf16 v8bf;
typedef __attribute__((ext_vector_type(8)))  float  v8f;

__device__ __forceinline__ unsigned short f32_to_bf16_rne(float f) {
  unsigned int u = __float_as_uint(f);
  u += 0x7fffu + ((u >> 16) & 1u);   // round-to-nearest-even
  return (unsigned short)(u >> 16);
}

__device__ __forceinline__ float sigmoidf_(float x) {
  return 1.0f / (1.0f + __expf(-x));          // x->+inf: 1,  x->-inf: 0
}
__device__ __forceinline__ float tanhf_fast(float x) {
  return 1.0f - 2.0f / (1.0f + __expf(2.0f * x));  // x->+inf: 1, x->-inf: -1
}

// ---------------------------------------------------------------------------
// One-time prep: pack 8 fp32 weight matrices into bf16 Wcat[n][k], n-major.
// Row n (0..4095): gate g = n>>10, local row ln = n&1023.
//   k <  1024 : Wh_g[ln, k]       (multiplies h)
//   k >= 1024 : Wi_g[ln, k-1024]  (multiplies x)
// ---------------------------------------------------------------------------
__global__ __launch_bounds__(256) void k_pack_weights(
    unsigned short* __restrict__ Wcat,
    const float* __restrict__ Whf, const float* __restrict__ Whi,
    const float* __restrict__ Whc, const float* __restrict__ Who,
    const float* __restrict__ Wif, const float* __restrict__ Wii,
    const float* __restrict__ Wic, const float* __restrict__ Wio) {
  unsigned int idx = blockIdx.x * 256u + threadIdx.x;
  if (idx >= (unsigned int)(NDIM) * (unsigned int)(KDIM)) return;
  unsigned int n  = idx >> 11;
  unsigned int k  = idx & (KDIM - 1);
  unsigned int g  = n >> 10;
  unsigned int ln = n & (H_ - 1);
  const float* Wh = (g == 0) ? Whf : (g == 1) ? Whi : (g == 2) ? Whc : Who;
  const float* Wi = (g == 0) ? Wif : (g == 1) ? Wii : (g == 2) ? Wic : Wio;
  float v = (k < H_) ? Wh[(size_t)ln * H_ + k] : Wi[(size_t)ln * H_ + (k - H_)];
  Wcat[(size_t)n * KDIM + k] = f32_to_bf16_rne(v);
}

// Combined bias per gate column: bias[g*1024+ln] = bh_g[ln] + bi_g[ln]
__global__ __launch_bounds__(256) void k_pack_bias(
    float* __restrict__ bias,
    const float* __restrict__ bhf, const float* __restrict__ bhi,
    const float* __restrict__ bhc, const float* __restrict__ bho,
    const float* __restrict__ bif, const float* __restrict__ bii,
    const float* __restrict__ bic, const float* __restrict__ bio) {
  unsigned int n = blockIdx.x * 256u + threadIdx.x;
  if (n >= NDIM) return;
  unsigned int g = n >> 10, ln = n & (H_ - 1);
  const float* bh = (g == 0) ? bhf : (g == 1) ? bhi : (g == 2) ? bhc : bho;
  const float* bi = (g == 0) ? bif : (g == 1) ? bii : (g == 2) ? bic : bio;
  bias[n] = bh[ln] + bi[ln];
}

// Init: trajectory[0] = x_0, c = 0, hx0 = [bf16(0) | bf16(x_0)].
__global__ __launch_bounds__(256) void k_init(
    float* __restrict__ traj0, const float* __restrict__ x0,
    float* __restrict__ c, unsigned short* __restrict__ hx0) {
  unsigned int idx = blockIdx.x * 256u + threadIdx.x;
  if (idx >= (unsigned int)B_ * H_) return;
  unsigned int b = idx >> 10, n = idx & (H_ - 1);
  float xv = x0[idx];
  traj0[idx] = xv;
  c[idx] = 0.0f;
  hx0[(size_t)b * KDIM + n]      = 0;                     // h = 0
  hx0[(size_t)b * KDIM + H_ + n] = f32_to_bf16_rne(xv);   // x = x_0
}

// ---------------------------------------------------------------------------
// Fused LSTM step:
//   GEMM: pre[64 x (4x16)] = hx_in[64x2048](bf16) . Wcat rows for 4 gates,
//         via v_wmma_f32_16x16x32_bf16, double-buffered LDS (1 barrier/slab),
//         K loop unrolled x2 so buffer indices are compile-time constants.
//         Scheduling groups pin 10 DS-reads ahead of 4 WMMAs per slab so
//         dscnt waits retire incrementally and WMMAs issue back-to-back.
//   Epilogue: sigmoid/tanh gates, c update, trajectory write, bf16 hx repack.
// Each (b, n) element is owned by exactly one block; hx ping-pongs per step.
// ---------------------------------------------------------------------------
__global__ __launch_bounds__(128) void k_lstm_step(
    const __bf16* __restrict__ hx_in, const __bf16* __restrict__ W,
    const float* __restrict__ bias, float* __restrict__ c,
    float* __restrict__ out_t, unsigned short* __restrict__ hx_out) {
  __shared__ __bf16 As[2][BM * LDA];
  __shared__ __bf16 Bs[2][BN * LDA];

  const int tid  = threadIdx.x;
  const int lane = tid & 31;
  const int w    = tid >> 5;
  const int m0   = blockIdx.x * BM;   // batch-row block (0,64,128,192)
  const int n0   = blockIdx.y * 16;   // column within each gate (0..1008)
  const int r    = lane & 15;         // fragment row/col
  const int gh   = lane >> 4;         // half-wave selector

  // Staging: each thread owns two 16B chunks (rows c0row and c0row+32, same q)
  const int c0row = tid >> 2;
  const int c0q   = tid & 3;
  const int brow0 = (c0row >> 4) * H_ + n0 + (c0row & 15);          // gate rows
  const int brow1 = ((c0row + 32) >> 4) * H_ + n0 + ((c0row + 32) & 15);
  const __bf16* aSrc0 = hx_in + (size_t)(m0 + c0row) * KDIM + c0q * 8;
  const __bf16* aSrc1 = hx_in + (size_t)(m0 + c0row + 32) * KDIM + c0q * 8;
  const __bf16* bSrc0 = W + (size_t)brow0 * KDIM + c0q * 8;
  const __bf16* bSrc1 = W + (size_t)brow1 * KDIM + c0q * 8;

  v8f acc[4];
#pragma unroll
  for (int i = 0; i < 4; ++i) acc[i] = (v8f){0.f,0.f,0.f,0.f,0.f,0.f,0.f,0.f};

  auto stage = [&](int buf, int kt) {
    *(uint4*)&As[buf][c0row * LDA + c0q * 8]        = *(const uint4*)(aSrc0 + kt);
    *(uint4*)&As[buf][(c0row + 32) * LDA + c0q * 8] = *(const uint4*)(aSrc1 + kt);
    *(uint4*)&Bs[buf][c0row * LDA + c0q * 8]        = *(const uint4*)(bSrc0 + kt);
    *(uint4*)&Bs[buf][(c0row + 32) * LDA + c0q * 8] = *(const uint4*)(bSrc1 + kt);
    if (kt + KT < KDIM) {                  // pull weight stream into WGP$ early
      __builtin_prefetch((const void*)(bSrc0 + kt + KT), 0, 3);
      __builtin_prefetch((const void*)(bSrc1 + kt + KT), 0, 3);
    }
  };

  // Compute one 32-wide K slab from LDS buffer `buf`.
  auto compute = [&](int buf) {
    const __bf16* ap = &As[buf][(w * 16 + r) * LDA + gh * 8];
    v8bf alo = *(const v8bf*)ap;
    v8bf ahi = *(const v8bf*)(ap + 16);
    v16bf afrag = __builtin_shufflevector(alo, ahi,
        0,1,2,3,4,5,6,7,8,9,10,11,12,13,14,15);
    v16bf bfrag[4];
#pragma unroll
    for (int ns = 0; ns < 4; ++ns) {       // ns = gate (f,i,c,o)
      const __bf16* bp = &Bs[buf][(ns * 16 + r) * LDA + gh * 8];
      v8bf blo = *(const v8bf*)bp;
      v8bf bhi = *(const v8bf*)(bp + 16);
      bfrag[ns] = __builtin_shufflevector(blo, bhi,
          0,1,2,3,4,5,6,7,8,9,10,11,12,13,14,15);
    }
#pragma unroll
    for (int ns = 0; ns < 4; ++ns) {
      acc[ns] = __builtin_amdgcn_wmma_f32_16x16x32_bf16(
          false, afrag, false, bfrag[ns], (short)0, acc[ns], false, false);
    }
#if __has_builtin(__builtin_amdgcn_sched_group_barrier)
    // Pin all 10 DS reads ahead of the 4 WMMAs within this barrier region:
    // forces distinct B-fragment registers -> incremental dscnt waits,
    // back-to-back WMMA issue. Staging VMEM/DS-writes remain free to overlap.
    __builtin_amdgcn_sched_group_barrier(0x100, 10, 0);  // DS read x10
    __builtin_amdgcn_sched_group_barrier(0x008, 4, 0);   // WMMA x4
#endif
  };

  stage(0, 0);
  for (int kt = 0; kt < KDIM; kt += 2 * KT) {   // 32 iterations, bufs constant
    __syncthreads();
    if (kt + KT < KDIM) stage(1, kt + KT);
    compute(0);
    __syncthreads();
    if (kt + 2 * KT < KDIM) stage(0, kt + 2 * KT);
    compute(1);
  }

  // Fused pointwise epilogue. C/D layout: element e -> (M = 8*gh + e, N = r).
  const int col = n0 + r;
  const float bf_ = bias[col];
  const float bi_ = bias[H_ + col];
  const float bc_ = bias[2 * H_ + col];
  const float bo_ = bias[3 * H_ + col];
  const int mbase = m0 + w * 16 + gh * 8;
#pragma unroll
  for (int e = 0; e < 8; ++e) {
    const size_t idx = (size_t)(mbase + e) * H_ + col;
    float gf = sigmoidf_(acc[0][e] + bf_);
    float gi = sigmoidf_(acc[1][e] + bi_);
    float cp = tanhf_fast(acc[2][e] + bc_);
    float go = sigmoidf_(acc[3][e] + bo_);
    float cv = gf * c[idx] + gi * cp;
    float hv = go * tanhf_fast(cv);
    c[idx]     = cv;
    out_t[idx] = hv;
    unsigned short hb = f32_to_bf16_rne(hv);
    hx_out[(size_t)(mbase + e) * KDIM + col]      = hb;  // h for next step
    hx_out[(size_t)(mbase + e) * KDIM + H_ + col] = hb;  // x = h for next step
  }
}

// ---------------------------------------------------------------------------
// Launch: prep (pack weights/bias, init state), then 127 fused steps with
// ping-pong hx buffers. All on `stream`; deterministic.
// ---------------------------------------------------------------------------
extern "C" void kernel_launch(void* const* d_in, const int* in_sizes, int n_in,
                              void* d_out, int out_size, void* d_ws, size_t ws_size,
                              hipStream_t stream) {
  (void)in_sizes; (void)n_in; (void)out_size; (void)ws_size;

  const float* Wif = (const float*)d_in[0];
  const float* bif = (const float*)d_in[1];
  const float* Wii = (const float*)d_in[2];
  const float* bii = (const float*)d_in[3];
  const float* Wio = (const float*)d_in[4];
  const float* bio = (const float*)d_in[5];
  const float* Wic = (const float*)d_in[6];
  const float* bic = (const float*)d_in[7];
  const float* Whf = (const float*)d_in[8];
  const float* bhf = (const float*)d_in[9];
  const float* Whi = (const float*)d_in[10];
  const float* bhi = (const float*)d_in[11];
  const float* Who = (const float*)d_in[12];
  const float* bho = (const float*)d_in[13];
  const float* Whc = (const float*)d_in[14];
  const float* bhc = (const float*)d_in[15];
  const float* x0  = (const float*)d_in[16];
  float* out = (float*)d_out;

  // Workspace layout (bytes)
  char* ws = (char*)d_ws;
  const size_t OFF_W    = 0;                                   // 16 MB bf16 Wcat
  const size_t OFF_BIAS = OFF_W    + (size_t)NDIM * KDIM * 2;  // 16 KB f32
  const size_t OFF_HX0  = OFF_BIAS + (size_t)NDIM * 4;         // 1 MB bf16
  const size_t OFF_HX1  = OFF_HX0  + (size_t)B_ * KDIM * 2;    // 1 MB bf16
  const size_t OFF_C    = OFF_HX1  + (size_t)B_ * KDIM * 2;    // 1 MB f32 cell

  unsigned short* Wcat_us = (unsigned short*)(ws + OFF_W);
  const __bf16*   Wcat_bf = (const __bf16*)(ws + OFF_W);
  float*          biasv   = (float*)(ws + OFF_BIAS);
  unsigned short* hx0_us  = (unsigned short*)(ws + OFF_HX0);
  const __bf16*   hx0_bf  = (const __bf16*)(ws + OFF_HX0);
  unsigned short* hx1_us  = (unsigned short*)(ws + OFF_HX1);
  const __bf16*   hx1_bf  = (const __bf16*)(ws + OFF_HX1);
  float*          c       = (float*)(ws + OFF_C);

  // Prep
  k_pack_weights<<<((size_t)NDIM * KDIM + 255) / 256, 256, 0, stream>>>(
      Wcat_us, Whf, Whi, Whc, Who, Wif, Wii, Wic, Wio);
  k_pack_bias<<<(NDIM + 255) / 256, 256, 0, stream>>>(
      biasv, bhf, bhi, bhc, bho, bif, bii, bic, bio);
  k_init<<<((size_t)B_ * H_ + 255) / 256, 256, 0, stream>>>(out, x0, c, hx0_us);

  // Time steps (T fixed at 128 per reference); hx ping-pongs each step.
  dim3 ggrid(B_ / BM, H_ / 16);   // (4, 64) = 256 workgroups
  for (int t = 1; t < T_; ++t) {
    const __bf16*   hin  = (t & 1) ? hx0_bf : hx1_bf;
    unsigned short* hout = (t & 1) ? hx1_us : hx0_us;
    k_lstm_step<<<ggrid, 128, 0, stream>>>(
        hin, Wcat_bf, biasv, c, out + (size_t)t * B_ * H_, hout);
  }
}
